// Attention_9955734192530
// MI455X (gfx1250) — compile-verified, matrix operand order checked by reading, not run on hardware
//
#include <hip/hip_runtime.h>

// ---------------------------------------------------------------------------
// Attention forward for MI455X (gfx1250): f16 WMMA everywhere, fp32 accum.
//   x:[B,T,D] f32, W_qkv:[3D,D], b_qkv:[3D], W_o:[D,D], b_o:[D]  ->  out:[B,T,D] f32
// ---------------------------------------------------------------------------

typedef _Float16 half8 __attribute__((ext_vector_type(8)));
typedef _Float16 v16h  __attribute__((ext_vector_type(16)));
typedef float    v8f   __attribute__((ext_vector_type(8)));

#define WMMA16(a, b, c) \
  __builtin_amdgcn_wmma_f32_16x16x32_f16(false, (a), false, (b), (short)0, (c), false, false)

constexpr int kB  = 2;
constexpr int kT  = 2048;
constexpr int kD  = 1024;
constexpr int kH  = 16;
constexpr int kDH = 64;
constexpr int kM  = kB * kT;   // 4096 rows of x
constexpr int kN1 = 3 * kD;    // 3072 qkv outputs

// Workspace layout (bytes). Total = 48 MiB.
constexpr size_t OFF_XH   = 0;                                    // f16 [4096,1024]  8 MiB
constexpr size_t OFF_WQKV = OFF_XH   + (size_t)kM  * kD * 2;      // f16 [3072,1024]  6 MiB
constexpr size_t OFF_WO   = OFF_WQKV + (size_t)kN1 * kD * 2;      // f16 [1024,1024]  2 MiB
constexpr size_t OFF_Q    = OFF_WO   + (size_t)kD  * kD * 2;      // f16 [B,H,T,DH]   8 MiB
constexpr size_t OFF_K    = OFF_Q    + (size_t)kM  * kD * 2 / kH * kH; // 8 MiB (== kB*kH*kT*kDH*2)
constexpr size_t OFF_V    = OFF_K    + (size_t)kB * kH * kT * kDH * 2; // f16 [B,H,DH,T] 8 MiB
constexpr size_t OFF_CTX  = OFF_V    + (size_t)kB * kH * kT * kDH * 2; // f16 [4096,1024] 8 MiB

// Load one 16x32 f16 A/B fragment slice for this lane.
// p must already point at:  row_base + (lane>=16 ? 8 : 0)
// Lanes 0-15 get K {0..7, 16..23}; lanes 16-31 get K {8..15, 24..31}.
__device__ __forceinline__ v16h ldfrag(const _Float16* p) {
  half8 lo = *reinterpret_cast<const half8*>(p);
  half8 hi = *reinterpret_cast<const half8*>(p + 16);
  v16h r;
#pragma unroll
  for (int i = 0; i < 8; ++i) { r[i] = lo[i]; r[i + 8] = hi[i]; }
  return r;
}

// ---------------------------------------------------------------------------
// Kernel 0: f32 -> f16 conversion
// ---------------------------------------------------------------------------
__global__ __launch_bounds__(256) void cvt_f32_f16(const float* __restrict__ src,
                                                   _Float16* __restrict__ dst, int n) {
  int i = blockIdx.x * 256 + threadIdx.x;
  if (i < n) dst[i] = (_Float16)src[i];
}

// ---------------------------------------------------------------------------
// Kernel 1: QKV GEMM.  qkv = x @ W_qkv^T + b_qkv, scattered to Q,K ([B,H,T,DH])
// and V transposed ([B,H,DH,T]).  One wave = 64x64 output tile.
// ---------------------------------------------------------------------------
__global__ __launch_bounds__(256) void qkv_gemm(const _Float16* __restrict__ xh,
                                                const _Float16* __restrict__ wh,
                                                const float* __restrict__ bias,
                                                _Float16* __restrict__ Qh,
                                                _Float16* __restrict__ Kh,
                                                _Float16* __restrict__ Vt) {
  const int lane = threadIdx.x & 31;
  const int wv   = threadIdx.x >> 5;
  const int wt   = blockIdx.x * 8 + wv;          // 0 .. 3071
  const int tm   = wt % (kM / 64);               // 64 row tiles
  const int tn   = wt / (kM / 64);               // 48 col tiles
  const int m0   = tm * 64;
  const int n0   = tn * 64;
  const int ml   = lane & 15;
  const int hi   = lane >> 4;

  v8f c[4][4] = {};
  for (int kk = 0; kk < kD; kk += 32) {
    v16h a[4], b[4];
#pragma unroll
    for (int mi = 0; mi < 4; ++mi)
      a[mi] = ldfrag(xh + (size_t)(m0 + mi * 16 + ml) * kD + kk + hi * 8);
#pragma unroll
    for (int ni = 0; ni < 4; ++ni)
      b[ni] = ldfrag(wh + (size_t)(n0 + ni * 16 + ml) * kD + kk + hi * 8);
#pragma unroll
    for (int mi = 0; mi < 4; ++mi)
#pragma unroll
      for (int ni = 0; ni < 4; ++ni)
        c[mi][ni] = WMMA16(a[mi], b[ni], c[mi][ni]);
  }

  // Epilogue: add bias, convert to f16, scatter into Q/K/V layouts.
#pragma unroll
  for (int ni = 0; ni < 4; ++ni) {
    const int n    = n0 + ni * 16 + ml;          // qkv column 0..3071
    const int mat  = n / kD;                     // 0=Q 1=K 2=V (uniform per tile)
    const int dcol = n % kD;
    const int h    = dcol >> 6;
    const int dh   = dcol & 63;
    const float bv = bias[n];
#pragma unroll
    for (int mi = 0; mi < 4; ++mi) {
#pragma unroll
      for (int g = 0; g < 8; ++g) {
        const int m  = m0 + mi * 16 + hi * 8 + g;
        const int bi = m >> 11;                  // batch
        const int t  = m & (kT - 1);
        const _Float16 hv = (_Float16)(c[mi][ni][g] + bv);
        const size_t bhh = (size_t)(bi * kH + h);
        if (mat == 0)      Qh[(bhh * kT + t) * kDH + dh] = hv;
        else if (mat == 1) Kh[(bhh * kT + t) * kDH + dh] = hv;
        else               Vt[(bhh * kDH + dh) * kT + t] = hv;
      }
    }
  }
}

// ---------------------------------------------------------------------------
// Kernel 2: flash attention. One wave = one (b,h) and one block of 16 queries.
// Streams keys in superblocks of 32; online softmax; P staged via LDS to
// convert C-layout -> A-fragment for the P @ V WMMAs.
// ---------------------------------------------------------------------------
__global__ __launch_bounds__(256) void attn_fwd(const _Float16* __restrict__ Qh,
                                                const _Float16* __restrict__ Kh,
                                                const _Float16* __restrict__ Vt,
                                                _Float16* __restrict__ ctx) {
  __shared__ _Float16 Pbuf[8][16 * 32];          // 1 KiB per wave
  const int lane = threadIdx.x & 31;
  const int wv   = threadIdx.x >> 5;
  const int wid  = blockIdx.x * 8 + wv;          // 0 .. 4095
  const int bh   = wid >> 7;                     // 0 .. 31  (b*16 + h)
  const int qb   = wid & 127;                    // query block of 16
  const int ml   = lane & 15;
  const int hi   = lane >> 4;
  const float scale = 0.125f;                    // 1/sqrt(64)

  const _Float16* Qb = Qh + ((size_t)bh * kT + qb * 16) * kDH;
  const _Float16* Kb = Kh + (size_t)bh * kT * kDH;
  const _Float16* Vb = Vt + (size_t)bh * kDH * kT;
  _Float16* P = &Pbuf[wv][0];

  const v16h qa0 = ldfrag(Qb + (size_t)ml * kDH + hi * 8);        // dh 0..31
  const v16h qa1 = ldfrag(Qb + (size_t)ml * kDH + 32 + hi * 8);   // dh 32..63

  float rm[8], rs[8];
  v8f o[4] = {};
#pragma unroll
  for (int g = 0; g < 8; ++g) { rm[g] = -3.0e38f; rs[g] = 0.0f; }

  for (int kb = 0; kb < kT; kb += 32) {
    // ---- S = scale * Q K^T for 16 queries x 32 keys (two 16x16 tiles) ----
    const _Float16* k0p = Kb + (size_t)(kb + ml)      * kDH + hi * 8;
    const _Float16* k1p = Kb + (size_t)(kb + 16 + ml) * kDH + hi * 8;
    v8f s0 = {}, s1 = {};
    s0 = WMMA16(qa0, ldfrag(k0p),      s0);
    s0 = WMMA16(qa1, ldfrag(k0p + 32), s0);
    s1 = WMMA16(qa0, ldfrag(k1p),      s1);
    s1 = WMMA16(qa1, ldfrag(k1p + 32), s1);

    // ---- online softmax (rows live inside 16-lane halves) ----
    float e0[8], e1[8];
#pragma unroll
    for (int g = 0; g < 8; ++g) {
      e0[g] = s0[g] * scale;
      e1[g] = s1[g] * scale;
      float t = fmaxf(e0[g], e1[g]);
      t = fmaxf(t, __shfl_xor(t, 1));
      t = fmaxf(t, __shfl_xor(t, 2));
      t = fmaxf(t, __shfl_xor(t, 4));
      t = fmaxf(t, __shfl_xor(t, 8));
      const float nm  = fmaxf(rm[g], t);
      const float scl = __expf(rm[g] - nm);
      rm[g] = nm;
      e0[g] = __expf(e0[g] - nm);
      e1[g] = __expf(e1[g] - nm);
      float sum = e0[g] + e1[g];
      sum += __shfl_xor(sum, 1);
      sum += __shfl_xor(sum, 2);
      sum += __shfl_xor(sum, 4);
      sum += __shfl_xor(sum, 8);
      rs[g] = rs[g] * scl + sum;
#pragma unroll
      for (int j = 0; j < 4; ++j) o[j][g] *= scl;
    }

    // ---- C-layout P -> LDS (row-major 16x32) -> A-fragment ----
#pragma unroll
    for (int g = 0; g < 8; ++g) {
      P[(hi * 8 + g) * 32 + ml]      = (_Float16)e0[g];
      P[(hi * 8 + g) * 32 + 16 + ml] = (_Float16)e1[g];
    }
    asm volatile("s_wait_dscnt 0" ::: "memory");
    const v16h pa = ldfrag(P + ml * 32 + hi * 8);

    // ---- O += P @ V  (V pre-transposed: rows are dh, contiguous in t) ----
#pragma unroll
    for (int j = 0; j < 4; ++j) {
      const _Float16* vp = Vb + (size_t)(j * 16 + ml) * kT + kb + hi * 8;
      o[j] = WMMA16(pa, ldfrag(vp), o[j]);
    }
  }

  // ---- normalize and write ctx [B*T, D] as f16 ----
  const int b = bh >> 4;
  const int h = bh & 15;
#pragma unroll
  for (int j = 0; j < 4; ++j) {
#pragma unroll
    for (int g = 0; g < 8; ++g) {
      const int t = qb * 16 + hi * 8 + g;
      const int d = h * 64 + j * 16 + ml;
      ctx[((size_t)(b * kT + t)) * kD + d] = (_Float16)(o[j][g] / rs[g]);
    }
  }
}

// ---------------------------------------------------------------------------
// Kernel 3: output projection.  out = ctx @ W_o^T + b_o  (f32 out)
// ---------------------------------------------------------------------------
__global__ __launch_bounds__(256) void out_gemm(const _Float16* __restrict__ ctx,
                                                const _Float16* __restrict__ wo,
                                                const float* __restrict__ bo,
                                                float* __restrict__ out) {
  const int lane = threadIdx.x & 31;
  const int wv   = threadIdx.x >> 5;
  const int wt   = blockIdx.x * 8 + wv;          // 0 .. 1023
  const int tm   = wt % (kM / 64);
  const int tn   = wt / (kM / 64);               // 0 .. 15
  const int m0   = tm * 64;
  const int n0   = tn * 64;
  const int ml   = lane & 15;
  const int hi   = lane >> 4;

  v8f c[4][4] = {};
  for (int kk = 0; kk < kD; kk += 32) {
    v16h a[4], b[4];
#pragma unroll
    for (int mi = 0; mi < 4; ++mi)
      a[mi] = ldfrag(ctx + (size_t)(m0 + mi * 16 + ml) * kD + kk + hi * 8);
#pragma unroll
    for (int ni = 0; ni < 4; ++ni)
      b[ni] = ldfrag(wo + (size_t)(n0 + ni * 16 + ml) * kD + kk + hi * 8);
#pragma unroll
    for (int mi = 0; mi < 4; ++mi)
#pragma unroll
      for (int ni = 0; ni < 4; ++ni)
        c[mi][ni] = WMMA16(a[mi], b[ni], c[mi][ni]);
  }

#pragma unroll
  for (int ni = 0; ni < 4; ++ni) {
    const int n = n0 + ni * 16 + ml;
    const float bv = bo[n];
#pragma unroll
    for (int mi = 0; mi < 4; ++mi) {
#pragma unroll
      for (int g = 0; g < 8; ++g) {
        const int m = m0 + mi * 16 + hi * 8 + g;
        out[(size_t)m * kD + n] = c[mi][ni][g] + bv;
      }
    }
  }
}

// ---------------------------------------------------------------------------
extern "C" void kernel_launch(void* const* d_in, const int* in_sizes, int n_in,
                              void* d_out, int out_size, void* d_ws, size_t ws_size,
                              hipStream_t stream) {
  const float* x     = (const float*)d_in[0];
  const float* Wqkv  = (const float*)d_in[1];
  const float* bqkv  = (const float*)d_in[2];
  const float* Wo    = (const float*)d_in[3];
  const float* bo    = (const float*)d_in[4];

  char* ws = (char*)d_ws;
  _Float16* xh    = (_Float16*)(ws + OFF_XH);
  _Float16* wqkvh = (_Float16*)(ws + OFF_WQKV);
  _Float16* woh   = (_Float16*)(ws + OFF_WO);
  _Float16* Qh    = (_Float16*)(ws + OFF_Q);
  _Float16* Kh    = (_Float16*)(ws + OFF_K);
  _Float16* Vt    = (_Float16*)(ws + OFF_V);
  _Float16* ctx   = (_Float16*)(ws + OFF_CTX);

  const int nx = kM * kD;          // 4,194,304
  const int nw = kN1 * kD;         // 3,145,728
  const int no = kD * kD;          // 1,048,576
  cvt_f32_f16<<<(nx + 255) / 256, 256, 0, stream>>>(x, xh, nx);
  cvt_f32_f16<<<(nw + 255) / 256, 256, 0, stream>>>(Wqkv, wqkvh, nw);
  cvt_f32_f16<<<(no + 255) / 256, 256, 0, stream>>>(Wo, woh, no);

  // (M/64)*(N/64) wave tiles / 8 waves per block
  qkv_gemm<<<(kM / 64) * (kN1 / 64) / 8, 256, 0, stream>>>(xh, wqkvh, bqkv, Qh, Kh, Vt);
  attn_fwd<<<(kB * kH * (kT / 16)) / 8, 256, 0, stream>>>(Qh, Kh, Vt, ctx);
  out_gemm<<<(kM / 64) * (kD / 64) / 8, 256, 0, stream>>>(ctx, woh, bo, (float*)d_out);
}
